// LSTM_60808146977081
// MI455X (gfx1250) — matrix operand order repaired and checked by reading, not compiled, not run on hardware
//
#include <hip/hip_runtime.h>

// Fused LSTM forward for B=4096, T=512, I=10, H=32 on gfx1250 (MI455X).
// One wave = 16 batch rows; weights resident as WMMA B-fragments;
// recurrence uses v_wmma_f32_16x16x32_bf16 (K=32 == H), fp32 accumulation.
// Per step: x-WMMAs issue first (overlapping the LDS h-relayout latency),
// h-WMMAs accumulate on top; bias folded into activation FMAs; native tanh.

#define TT 512
#define II 10
#define HH 32

typedef __attribute__((ext_vector_type(16))) __bf16 bf16x16;
typedef __attribute__((ext_vector_type(8)))  float  v8f;

#if __has_builtin(__builtin_amdgcn_tanhf)
  #define TANHF(x) __builtin_amdgcn_tanhf(x)
#elif __has_builtin(__builtin_amdgcn_tanh_f32)
  #define TANHF(x) __builtin_amdgcn_tanh_f32(x)
#else
  __device__ __forceinline__ float TANHF(float v) {
      return 2.f / (1.f + __expf(-2.f * v)) - 1.f;
  }
#endif

__global__ __launch_bounds__(64, 1)
__attribute__((amdgpu_waves_per_eu(4)))
void lstm_fused_wmma(
    const float* __restrict__ x,     // [B, T, I]
    const float* __restrict__ h0,    // [B, H]
    const float* __restrict__ c0,    // [B, H]
    const float* __restrict__ W_ih,  // [4H, I]
    const float* __restrict__ W_hh,  // [4H, H]
    const float* __restrict__ b_ih,  // [4H]
    const float* __restrict__ b_hh,  // [4H]
    const float* __restrict__ W_lin, // [1, H]
    const float* __restrict__ b_lin, // [1]
    float* __restrict__ out)         // [B, 1]
{
    const int lane = threadIdx.x & 31;
    const int wave = threadIdx.x >> 5;   // 2 waves per block
    const int lgrp = lane >> 4;          // 0 = lanes 0-15, 1 = lanes 16-31
    const int lm   = lane & 15;
    const int b0   = blockIdx.x * 32 + wave * 16;   // first batch row of wave

    // Per-wave h staging (bf16 row-major) for C-layout -> A-layout relayout.
    __shared__ __bf16 h_s[2][16][HH];

    // ---- Resident weight fragments ---------------------------------------
    // B-fragment (32x16 KxN) tile n0: B[k][n] = W[(16*n0+n)*K + k];
    // lane holds column n = lm, K half selected by lgrp.
    bf16x16 bhh[8];
#pragma unroll
    for (int n0 = 0; n0 < 8; ++n0) {
        const float* wrow = W_hh + (size_t)(16 * n0 + lm) * HH + 16 * lgrp;
        bf16x16 v;
#pragma unroll
        for (int e = 0; e < 16; ++e) v[e] = (__bf16)wrow[e];
        bhh[n0] = v;
    }
    bf16x16 bih[8];   // K padded 10 -> 32 with zeros
#pragma unroll
    for (int n0 = 0; n0 < 8; ++n0) {
        const float* wrow = W_ih + (size_t)(16 * n0 + lm) * II;
        bf16x16 v;
#pragma unroll
        for (int e = 0; e < 16; ++e) {
            const int k = (e < 8) ? 8 * lgrp + e : 8 + 8 * lgrp + e;
            v[e] = (k < II) ? (__bf16)wrow[k] : (__bf16)0.f;
        }
        bih[n0] = v;
    }

    // Per-lane folded bias for gate column g = 16*n0 + lm.
    // sigmoid tiles (0..3, 6,7): 0.5*bias (sigmoid(x)=0.5*tanh(0.5x+0.5b)+0.5);
    // tanh tiles (4,5): full bias.
    float hb[8];
#pragma unroll
    for (int n0 = 0; n0 < 8; ++n0) {
        const int g = 16 * n0 + lm;
        const float b = b_ih[g] + b_hh[g];
        hb[n0] = (n0 == 4 || n0 == 5) ? b : 0.5f * b;
    }

    // c state in C-matrix layout: value(n,j) = c[row j+8*lgrp][col 16*n+lm]
    float cst[2][8];
#pragma unroll
    for (int n = 0; n < 2; ++n)
#pragma unroll
        for (int j = 0; j < 8; ++j)
            cst[n][j] = c0[(size_t)(b0 + j + 8 * lgrp) * HH + 16 * n + lm];

    // h0 -> LDS (each wave touches only its own region)
    for (int idx = lane; idx < 16 * HH; idx += 32)
        h_s[wave][idx >> 5][idx & 31] =
            (__bf16)h0[(size_t)(b0 + (idx >> 5)) * HH + (idx & 31)];

    // ---- x pipeline: one step ahead in registers -------------------------
    const float* xrow = x + (size_t)(b0 + lm) * TT * II;   // this lane's row base
    float2 xc[4], xn[4];
    {
        const float* rp = xrow;
        if (lgrp == 0) {
            xc[0] = *(const float2*)(rp + 0); xc[1] = *(const float2*)(rp + 2);
            xc[2] = *(const float2*)(rp + 4); xc[3] = *(const float2*)(rp + 6);
        } else {
            xc[0] = *(const float2*)(rp + 8);
            xc[1] = make_float2(0.f, 0.f); xc[2] = xc[1]; xc[3] = xc[1];
        }
    }

    // ---- Recurrent loop ---------------------------------------------------
    for (int t = 0; t < TT; ++t) {
        // A-fragment of x for this step (K 0..9 valid, rest zero) — ready
        // before the LDS h-relayout completes.
        bf16x16 a_x;
#pragma unroll
        for (int p = 0; p < 4; ++p) {
            a_x[2 * p]     = (__bf16)xc[p].x;
            a_x[2 * p + 1] = (__bf16)xc[p].y;
        }
#pragma unroll
        for (int e = 8; e < 16; ++e) a_x[e] = (__bf16)0.f;

        // Kick off the LDS relayout loads for a_h (2x ds_load_b128).
        bf16x16 a_h;
        {
            const __bf16* row = &h_s[wave][lm][0];
#pragma unroll
            for (int p = 0; p < 8; ++p) {
                const int k = 2 * p + 16 * lgrp;
                a_h[2 * p]     = row[k];
                a_h[2 * p + 1] = row[k + 1];
            }
        }

        // Prefetch / pipeline next step's x while WMMAs run.
        if (t + 1 < TT) {
            const float* rp = xrow + (size_t)(t + 1) * II;
            if (lgrp == 0) {
                xn[0] = *(const float2*)(rp + 0); xn[1] = *(const float2*)(rp + 2);
                xn[2] = *(const float2*)(rp + 4); xn[3] = *(const float2*)(rp + 6);
            } else {
                xn[0] = *(const float2*)(rp + 8);
                xn[1] = make_float2(0.f, 0.f); xn[2] = xn[1]; xn[3] = xn[1];
            }
        }
        if (t + 8 < TT)   // deep prefetch to hide HBM latency in serial chain
            __builtin_prefetch(xrow + (size_t)(t + 8) * II, 0, 0);

        // Two groups of gate tiles: group n covers (i,f,g,o) = tiles n,n+2,n+4,n+6.
        // x-WMMAs first (independent of a_h, overlap DS latency); h-WMMAs
        // accumulate on top — only one WMMA sits on the recurrence chain.
#pragma unroll
        for (int n = 0; n < 2; ++n) {
            const v8f zc = {0.f, 0.f, 0.f, 0.f, 0.f, 0.f, 0.f, 0.f};
            v8f ai = __builtin_amdgcn_wmma_f32_16x16x32_bf16(
                false, a_x, false, bih[n    ], (short)0, zc, false, false);
            v8f af = __builtin_amdgcn_wmma_f32_16x16x32_bf16(
                false, a_x, false, bih[n + 2], (short)0, zc, false, false);
            v8f ag = __builtin_amdgcn_wmma_f32_16x16x32_bf16(
                false, a_x, false, bih[n + 4], (short)0, zc, false, false);
            v8f ao = __builtin_amdgcn_wmma_f32_16x16x32_bf16(
                false, a_x, false, bih[n + 6], (short)0, zc, false, false);
            ai = __builtin_amdgcn_wmma_f32_16x16x32_bf16(
                false, a_h, false, bhh[n    ], (short)0, ai, false, false);
            af = __builtin_amdgcn_wmma_f32_16x16x32_bf16(
                false, a_h, false, bhh[n + 2], (short)0, af, false, false);
            ag = __builtin_amdgcn_wmma_f32_16x16x32_bf16(
                false, a_h, false, bhh[n + 4], (short)0, ag, false, false);
            ao = __builtin_amdgcn_wmma_f32_16x16x32_bf16(
                false, a_h, false, bhh[n + 6], (short)0, ao, false, false);

            // Cell update; sigmoid(x) = 0.5*tanh(0.5x + 0.5b) + 0.5 (1 TRANS op)
#pragma unroll
            for (int j = 0; j < 8; ++j) {
                const float ig = fmaf(0.5f, TANHF(fmaf(0.5f, ai[j], hb[n    ])), 0.5f);
                const float fg = fmaf(0.5f, TANHF(fmaf(0.5f, af[j], hb[n + 2])), 0.5f);
                const float gg = TANHF(ag[j] + hb[n + 4]);
                const float og = fmaf(0.5f, TANHF(fmaf(0.5f, ao[j], hb[n + 6])), 0.5f);
                const float cn = fmaf(fg, cst[n][j], ig * gg);
                cst[n][j] = cn;
                h_s[wave][j + 8 * lgrp][16 * n + lm] = (__bf16)(og * TANHF(cn));
            }
        }

#pragma unroll
        for (int p = 0; p < 4; ++p) xc[p] = xn[p];
    }

    // ---- Final projection: out[b] = h_T[b,:] . W_lin + b_lin --------------
    if (lgrp == 0) {
        float a = b_lin[0];
#pragma unroll
        for (int k = 0; k < HH; ++k)
            a += (float)h_s[wave][lm][k] * W_lin[k];
        out[b0 + lm] = a;
    }
}

extern "C" void kernel_launch(void* const* d_in, const int* in_sizes, int n_in,
                              void* d_out, int out_size, void* d_ws, size_t ws_size,
                              hipStream_t stream) {
    const float* x     = (const float*)d_in[0];
    const float* h0    = (const float*)d_in[1];
    const float* c0    = (const float*)d_in[2];
    const float* W_ih  = (const float*)d_in[3];
    const float* W_hh  = (const float*)d_in[4];
    const float* b_ih  = (const float*)d_in[5];
    const float* b_hh  = (const float*)d_in[6];
    const float* W_lin = (const float*)d_in[7];
    const float* b_lin = (const float*)d_in[8];
    float* out = (float*)d_out;

    const int B = in_sizes[1] / HH;          // h0 is [B, H]
    dim3 grid(B / 32), block(64);            // 2 waves/block, 16 rows/wave
    lstm_fused_wmma<<<grid, block, 0, stream>>>(
        x, h0, c0, W_ih, W_hh, b_ih, b_hh, W_lin, b_lin, out);
}
